// SentenceDGConvNodeClassifier_70875550318665
// MI455X (gfx1250) — matrix operand order repaired and checked by reading, not compiled, not run on hardware
//
#include <hip/hip_runtime.h>
#include <hip/hip_bf16.h>

// ---------------------------------------------------------------------------
// Types for CDNA5 WMMA (wave32): 16x16x32 bf16 -> f32 accumulate
// ---------------------------------------------------------------------------
typedef __attribute__((ext_vector_type(16))) __bf16 v16bf;
typedef __attribute__((ext_vector_type(8)))  float  v8f;

__device__ __forceinline__ unsigned short f2bf(float f) {
  // round-to-nearest-even fp32 -> bf16 (bit-level, avoids __bf16 scalar ops)
  unsigned int u = __float_as_uint(f);
  unsigned int r = (u + 0x7FFFu + ((u >> 16) & 1u)) >> 16;
  return (unsigned short)r;
}

// ---------------------------------------------------------------------------
// Small elementwise / setup kernels
// ---------------------------------------------------------------------------
__global__ void k_fill_ones(float* v, int n) {
  int i = blockIdx.x * blockDim.x + threadIdx.x;
  if (i < n) v[i] = 1.0f;
}

__global__ void k_deg_edges(float* deg, const int* __restrict__ dst, int e) {
  int i = blockIdx.x * blockDim.x + threadIdx.x;
  if (i < e) atomicAdd(&deg[dst[i]], 1.0f);
}

__global__ void k_rsqrt_inplace(float* v, int n) {
  int i = blockIdx.x * blockDim.x + threadIdx.x;
  if (i < n) {
    float d = v[i];
    v[i] = (d > 0.0f) ? rsqrtf(d) : 0.0f;
  }
}

__global__ void k_f32_to_bf16(const float* __restrict__ in, unsigned short* __restrict__ out, int n) {
  int i = blockIdx.x * blockDim.x + threadIdx.x;
  if (i < n) out[i] = f2bf(in[i]);
}

// W[K,Nc] row-major fp32 -> Wt[Nc,K] row-major bf16
__global__ void k_transpose_bf16(const float* __restrict__ w, unsigned short* __restrict__ wt,
                                 int K, int Nc) {
  int i = blockIdx.x * blockDim.x + threadIdx.x;
  if (i < K * Nc) {
    int k = i / Nc, n = i % Nc;
    wt[(size_t)n * K + k] = f2bf(w[i]);
  }
}

// ---------------------------------------------------------------------------
// bf16 WMMA GEMM: C[M,Nc] = A[M,K] @ B[K,Nc]
//   A  : bf16 row-major [M,K]
//   Bt : bf16 row-major [Nc,K]  (B transposed so B-fragments load contiguously)
//   C  : fp32 row-major [M,Nc]
// One wave computes a 16x64 strip (4 n-subtiles), reusing the A fragment.
// Requires: M%16==0, K%32==0, Nc%64==0 (holds: M=20000, K in {768,256}, Nc=256).
//
// Fragment layouts used here (wave32; contiguous k for both A and B, not the ISA table order):
//   A 16x32 bf16: lane l -> row M=l&15; 16 contiguous K starting at (l>>4)*16
//                 (two contiguous 16B chunks)
//   B 32x16 bf16: lane l -> col N=l&15; 16 contiguous K starting at (l>>4)*16
//   C 16x16 f32 : VGPR r, lane l -> M=r+(l>>4)*8, N=l&15
// ---------------------------------------------------------------------------
__global__ void k_gemm_bf16_wmma(const unsigned short* __restrict__ A,
                                 const unsigned short* __restrict__ Bt,
                                 float* __restrict__ C,
                                 int M, int K, int Nc) {
  const int nGroups = Nc >> 6;                       // 64 columns per wave
  const int wavesPerBlock = blockDim.x >> 5;
  const int wave = blockIdx.x * wavesPerBlock + (threadIdx.x >> 5);
  const int mTiles = M >> 4;
  const int mTile = wave / nGroups;
  if (mTile >= mTiles) return;                       // wave-uniform: EXEC stays all-1s in live waves
  const int n0   = (wave % nGroups) << 6;
  const int lane = threadIdx.x & 31;
  const int lr   = lane & 15;                        // A row / B,C column within tile
  const int half = lane >> 4;

  const unsigned short* aRow  = A  + (size_t)(mTile * 16 + lr) * K;
  const unsigned short* bRow0 = Bt + (size_t)(n0 +  0 + lr) * K;
  const unsigned short* bRow1 = Bt + (size_t)(n0 + 16 + lr) * K;
  const unsigned short* bRow2 = Bt + (size_t)(n0 + 32 + lr) * K;
  const unsigned short* bRow3 = Bt + (size_t)(n0 + 48 + lr) * K;

  v8f acc0 = {}, acc1 = {}, acc2 = {}, acc3 = {};
  union Frag { v16bf v; uint4 u[2]; };

  for (int k0 = 0; k0 < K; k0 += 32) {
    Frag fa;
    fa.u[0] = *(const uint4*)(aRow + k0 + half * 16);        // elems 0..7
    fa.u[1] = *(const uint4*)(aRow + k0 + half * 16 + 8);    // elems 8..15

    const int kb = k0 + half * 16;
    Frag fb0, fb1, fb2, fb3;
    fb0.u[0] = *(const uint4*)(bRow0 + kb); fb0.u[1] = *(const uint4*)(bRow0 + kb + 8);
    fb1.u[0] = *(const uint4*)(bRow1 + kb); fb1.u[1] = *(const uint4*)(bRow1 + kb + 8);
    fb2.u[0] = *(const uint4*)(bRow2 + kb); fb2.u[1] = *(const uint4*)(bRow2 + kb + 8);
    fb3.u[0] = *(const uint4*)(bRow3 + kb); fb3.u[1] = *(const uint4*)(bRow3 + kb + 8);

    acc0 = __builtin_amdgcn_wmma_f32_16x16x32_bf16(false, fa.v, false, fb0.v, (short)0, acc0, false, false);
    acc1 = __builtin_amdgcn_wmma_f32_16x16x32_bf16(false, fa.v, false, fb1.v, (short)0, acc1, false, false);
    acc2 = __builtin_amdgcn_wmma_f32_16x16x32_bf16(false, fa.v, false, fb2.v, (short)0, acc2, false, false);
    acc3 = __builtin_amdgcn_wmma_f32_16x16x32_bf16(false, fa.v, false, fb3.v, (short)0, acc3, false, false);
  }

  float* cBase = C + (size_t)(mTile * 16 + half * 8) * Nc + n0 + lr;
#pragma unroll
  for (int r = 0; r < 8; ++r) {
    cBase[(size_t)r * Nc +  0] = acc0[r];
    cBase[(size_t)r * Nc + 16] = acc1[r];
    cBase[(size_t)r * Nc + 32] = acc2[r];
    cBase[(size_t)r * Nc + 48] = acc3[r];
  }
}

// ---------------------------------------------------------------------------
// GCN aggregation: agg[i] = sum_{j->i} norm * t[j]  (+ self-loop), then tanh+bias
// ---------------------------------------------------------------------------
__global__ void k_agg_init(const float* __restrict__ t, const float* __restrict__ dinv,
                           float* __restrict__ agg, int n, int h) {
  int i = blockIdx.x * blockDim.x + threadIdx.x;
  if (i < n * h) {
    int nd = i / h;
    float w = dinv[nd] * dinv[nd];           // self-loop norm
    agg[i] = w * t[i];
  }
}

__global__ void k_agg_edges(const float* __restrict__ t, const float* __restrict__ dinv,
                            const int* __restrict__ src, const int* __restrict__ dst,
                            float* __restrict__ agg, int e, int h) {
  int i = blockIdx.x * blockDim.x + threadIdx.x;
  if (i < e * h) {
    int ed = i / h, c = i - ed * h;
    int s = src[ed], d = dst[ed];
    float w = dinv[s] * dinv[d];
    atomicAdd(&agg[(size_t)d * h + c], w * t[(size_t)s * h + c]);
  }
}

// h = tanh(agg + b); write fp32 into node_features slab and bf16 staging
__global__ void k_tanh_store(const float* __restrict__ agg, const float* __restrict__ bias,
                             float* __restrict__ nf, unsigned short* __restrict__ hbf,
                             int n, int h, int F, int colOff) {
  int i = blockIdx.x * blockDim.x + threadIdx.x;
  if (i < n * h) {
    int nd = i / h, c = i - nd * h;
    float v = tanhf(agg[i] + bias[c]);
    nf[(size_t)nd * F + colOff + c] = v;
    hbf[i] = f2bf(v);
  }
}

// layer 3: t3[i] = dot(h2[i,:], W2[:,0]); h2 lives inside nf slab at colOff
__global__ void k_matvec(const float* __restrict__ nf, const float* __restrict__ w2,
                         float* __restrict__ t3, int n, int h, int F, int colOff) {
  int i = blockIdx.x * blockDim.x + threadIdx.x;
  if (i < n) {
    const float* row = nf + (size_t)i * F + colOff;
    float s = 0.0f;
    for (int j = 0; j < h; ++j) s += row[j] * w2[j];
    t3[i] = s;
  }
}

__global__ void k_agg3_init(const float* __restrict__ t3, const float* __restrict__ dinv,
                            float* __restrict__ agg3, int n) {
  int i = blockIdx.x * blockDim.x + threadIdx.x;
  if (i < n) agg3[i] = dinv[i] * dinv[i] * t3[i];
}

__global__ void k_agg3_edges(const float* __restrict__ t3, const float* __restrict__ dinv,
                             const int* __restrict__ src, const int* __restrict__ dst,
                             float* __restrict__ agg3, int e) {
  int i = blockIdx.x * blockDim.x + threadIdx.x;
  if (i < e) {
    int s = src[i], d = dst[i];
    atomicAdd(&agg3[d], dinv[s] * dinv[d] * t3[s]);
  }
}

__global__ void k_tanh3(const float* __restrict__ agg3, const float* __restrict__ b2,
                        float* __restrict__ nf, int n, int F) {
  int i = blockIdx.x * blockDim.x + threadIdx.x;
  if (i < n) nf[(size_t)i * F + (F - 1)] = tanhf(agg3[i] + b2[0]);
}

// logits[n,c] = nf[n,:] @ Wc[:,c] + bc[c]
__global__ void k_logits(const float* __restrict__ nf, const float* __restrict__ wc,
                         const float* __restrict__ bc, float* __restrict__ out,
                         int n, int F, int C) {
  int i = blockIdx.x * blockDim.x + threadIdx.x;
  if (i < n * C) {
    int nd = i / C, c = i - nd * C;
    const float* row = nf + (size_t)nd * F;
    float s = bc[c];
    for (int j = 0; j < F; ++j) s += row[j] * wc[(size_t)j * C + c];
    out[i] = s;
  }
}

// edge_pairs[e] = concat(nf[src[e]], nf[dst[e]])  — dominant bandwidth consumer
__global__ void k_edge_pairs(const float* __restrict__ nf, const int* __restrict__ src,
                             const int* __restrict__ dst, float* __restrict__ ep,
                             int e, int F) {
  int ed = blockIdx.x;
  if (ed >= e) return;
  const float* srow = nf + (size_t)src[ed] * F;
  const float* trow = nf + (size_t)dst[ed] * F;
  float* orow = ep + (size_t)ed * (2 * F);
  for (int j = threadIdx.x; j < F; j += blockDim.x) {
    orow[j]     = srow[j];
    orow[F + j] = trow[j];
  }
}

// ---------------------------------------------------------------------------
// Host launch
// ---------------------------------------------------------------------------
static inline int cdiv(long long a, int b) { return (int)((a + b - 1) / b); }

extern "C" void kernel_launch(void* const* d_in, const int* in_sizes, int n_in,
                              void* d_out, int out_size, void* d_ws, size_t ws_size,
                              hipStream_t stream) {
  (void)n_in; (void)out_size; (void)ws_size;

  const float* x  = (const float*)d_in[0];
  const int*   ei = (const int*)d_in[1];
  const float* W0 = (const float*)d_in[2];
  const float* b0 = (const float*)d_in[3];
  const float* W1 = (const float*)d_in[4];
  const float* b1 = (const float*)d_in[5];
  const float* W2 = (const float*)d_in[6];
  const float* b2 = (const float*)d_in[7];
  const float* Wc = (const float*)d_in[8];
  const float* bc = (const float*)d_in[9];

  const int H = in_sizes[3];           // 256
  const int D = in_sizes[2] / H;       // 768
  const int N = in_sizes[0] / D;       // 20000
  const int E = in_sizes[1] / 2;       // 160000
  const int C = in_sizes[9];           // 10
  const int F = 2 * H + 1;             // 513

  const int* src = ei;
  const int* dst = ei + E;

  // workspace carve (256B aligned)
  char* p = (char*)d_ws;
  auto carve = [&](size_t bytes) -> char* {
    char* r = p;
    p += (bytes + 255) & ~(size_t)255;
    return r;
  };
  float*          dinv = (float*)carve((size_t)N * 4);
  unsigned short* xbf  = (unsigned short*)carve((size_t)N * D * 2);
  unsigned short* w0t  = (unsigned short*)carve((size_t)D * H * 2);
  unsigned short* w1t  = (unsigned short*)carve((size_t)H * H * 2);
  float*          tbuf = (float*)carve((size_t)N * H * 4);
  float*          agg  = (float*)carve((size_t)N * H * 4);
  unsigned short* hbf  = (unsigned short*)carve((size_t)N * H * 2);
  float*          t3   = (float*)carve((size_t)N * 4);
  float*          agg3 = (float*)carve((size_t)N * 4);

  // output slabs: [logits | node_features | edge_pairs]
  float* logits = (float*)d_out;
  float* nf     = logits + (size_t)N * C;
  float* ep     = nf + (size_t)N * F;

  const int B = 256;

  // --- degrees / normalization ---
  k_fill_ones    <<<cdiv(N, B), B, 0, stream>>>(dinv, N);
  k_deg_edges    <<<cdiv(E, B), B, 0, stream>>>(dinv, dst, E);
  k_rsqrt_inplace<<<cdiv(N, B), B, 0, stream>>>(dinv, N);

  // --- bf16 staging for WMMA ---
  k_f32_to_bf16  <<<cdiv((long long)N * D, B), B, 0, stream>>>(x, xbf, N * D);
  k_transpose_bf16<<<cdiv((long long)D * H, B), B, 0, stream>>>(W0, w0t, D, H);
  k_transpose_bf16<<<cdiv((long long)H * H, B), B, 0, stream>>>(W1, w1t, H, H);

  const int wavesPerBlock = B / 32;
  const int gemmWaves = (N / 16) * (H / 64);
  const int gemmBlocks = cdiv(gemmWaves, wavesPerBlock);

  // --- layer 1: t = x @ W0 ; aggregate ; h1 = tanh(agg + b0) ---
  k_gemm_bf16_wmma<<<gemmBlocks, B, 0, stream>>>(xbf, w0t, tbuf, N, D, H);
  k_agg_init <<<cdiv((long long)N * H, B), B, 0, stream>>>(tbuf, dinv, agg, N, H);
  k_agg_edges<<<cdiv((long long)E * H, B), B, 0, stream>>>(tbuf, dinv, src, dst, agg, E, H);
  k_tanh_store<<<cdiv((long long)N * H, B), B, 0, stream>>>(agg, b0, nf, hbf, N, H, F, 0);

  // --- layer 2: t = h1 @ W1 ; aggregate ; h2 = tanh(agg + b1) ---
  k_gemm_bf16_wmma<<<gemmBlocks, B, 0, stream>>>(hbf, w1t, tbuf, N, H, H);
  k_agg_init <<<cdiv((long long)N * H, B), B, 0, stream>>>(tbuf, dinv, agg, N, H);
  k_agg_edges<<<cdiv((long long)E * H, B), B, 0, stream>>>(tbuf, dinv, src, dst, agg, E, H);
  k_tanh_store<<<cdiv((long long)N * H, B), B, 0, stream>>>(agg, b1, nf, hbf, N, H, F, H);

  // --- layer 3 (H -> 1): matvec ; aggregate ; h3 = tanh(agg + b2) ---
  k_matvec   <<<cdiv(N, B), B, 0, stream>>>(nf, W2, t3, N, H, F, H);
  k_agg3_init<<<cdiv(N, B), B, 0, stream>>>(t3, dinv, agg3, N);
  k_agg3_edges<<<cdiv(E, B), B, 0, stream>>>(t3, dinv, src, dst, agg3, E);
  k_tanh3    <<<cdiv(N, B), B, 0, stream>>>(agg3, b2, nf, N, F);

  // --- classifier + edge-pair gather ---
  k_logits    <<<cdiv((long long)N * C, B), B, 0, stream>>>(nf, Wc, bc, logits, N, F, C);
  k_edge_pairs<<<E, B, 0, stream>>>(nf, src, dst, ep, E, F);
}
